// DagEncoder_29188597743898
// MI455X (gfx1250) — matrix operand order repaired and checked by reading, not compiled, not run on hardware
//
#include <hip/hip_runtime.h>
#include <hip/hip_bf16.h>

typedef __attribute__((ext_vector_type(16))) _Float16 v16h;
typedef __attribute__((ext_vector_type(8)))  float    v8f;

#define BLOCK_THREADS 256
#define WAVES_PER_BLOCK 8
#define ROWS_PER_BLOCK (WAVES_PER_BLOCK * 16)
#define EDIM 128
#define KDIM 256

// Fused [concat -> linear -> segment_csr(sum)] for the DAG encoder.
// A (per 16-row tile) : 16x256 f32, split into f16 hi/lo in registers
// B (W^T)             : 256x128, staged once per block into LDS as f16 hi/lo
//                       in WMMA-B fragment layout (lane-contiguous 16 halfs)
// C                   : f32 accumulators, 3x wmma_f32_16x16x32_f16 per tile-K
__global__ __launch_bounds__(BLOCK_THREADS)
void dag_encoder_wmma_kernel(const float* __restrict__ h_node,
                             const float* __restrict__ x,
                             const long long* __restrict__ ptr,
                             const float* __restrict__ W,
                             const float* __restrict__ b,
                             float* __restrict__ out,
                             int nRows, int nSeg)
{
    // 8 ksteps * 8 ntiles * 32 lanes fragments, 16 halfs (32B) per lane
    __shared__ v16h sBhi[8 * 8 * 32];   // 64 KB
    __shared__ v16h sBlo[8 * 8 * 32];   // 64 KB

    const int tid  = threadIdx.x;
    const int lane = tid & 31;
    const int wave = tid >> 5;

    // ---- Stage W into LDS as hi/lo f16 B-fragments (once per block) ----
    // Fragment (ks, nt), lane fl: column n = nt*16 + (fl&15),
    //   K = ks*32 + (fl<16 ? 0..15 : 16..31)  -> 16 contiguous floats of W row n
    for (int i = tid; i < 8 * 8 * 32; i += BLOCK_THREADS) {
        const int fl = i & 31;
        const int nt = (i >> 5) & 7;
        const int ks = i >> 8;
        const int n  = nt * 16 + (fl & 15);
        const int kb = ks * 32 + ((fl >> 4) << 4);
        const float* wp = W + (size_t)n * KDIM + kb;
        v16h hi, lo;
#pragma unroll
        for (int j = 0; j < 16; ++j) {
            float w = wp[j];
            _Float16 h = (_Float16)w;
            hi[j] = h;
            lo[j] = (_Float16)(w - (float)h);
        }
        sBhi[i] = hi;
        sBlo[i] = lo;
    }
    __syncthreads();

    const int m0 = blockIdx.x * ROWS_PER_BLOCK + wave * 16;
    if (m0 >= nRows) return;   // whole-wave exit only (N % 16 == 0), no barriers follow

    // ---- A addressing: contiguous-k lane layout (same k order as sBhi/sBlo) ----
    // lanes 0-15 : row m0+lane,     K = ks*32 + {0..15}
    // lanes 16-31: row m0+lane-16,  K = ks*32 + {16..31}
    const int rowA = m0 + (lane & 15);
    const float* xrow = x      + (size_t)rowA * 128;
    const float* hrow = h_node + (size_t)rowA * 128;
    const int halfOff = (lane >> 4) << 4;   // 0 or 16

    v8f acc[8];
#pragma unroll
    for (int t = 0; t < 8; ++t) acc[t] = {};

    for (int ks = 0; ks < 8; ++ks) {
        // concat(x, h_node): ksteps 0..3 read x, 4..7 read h_node
        const float* src = (ks < 4) ? (xrow + ks * 32) : (hrow + (ks - 4) * 32);
        const float4 c0 = *(const float4*)(src + halfOff);
        const float4 c1 = *(const float4*)(src + halfOff + 4);
        const float4 c2 = *(const float4*)(src + halfOff + 8);
        const float4 c3 = *(const float4*)(src + halfOff + 12);
        const float af[16] = { c0.x, c0.y, c0.z, c0.w,  c1.x, c1.y, c1.z, c1.w,
                               c2.x, c2.y, c2.z, c2.w,  c3.x, c3.y, c3.z, c3.w };
        v16h ah, al;
#pragma unroll
        for (int j = 0; j < 16; ++j) {
            _Float16 h = (_Float16)af[j];
            ah[j] = h;
            al[j] = (_Float16)(af[j] - (float)h);
        }
#pragma unroll
        for (int nt = 0; nt < 8; ++nt) {
            const v16h bh = sBhi[(ks * 8 + nt) * 32 + lane];
            const v16h bl = sBlo[(ks * 8 + nt) * 32 + lane];
            // split-precision: A*B ~= Ahi*Bhi + Ahi*Blo + Alo*Bhi  (f32 accum)
            acc[nt] = __builtin_amdgcn_wmma_f32_16x16x32_f16(
                false, ah, false, bh, (short)0, acc[nt], false, false);
            acc[nt] = __builtin_amdgcn_wmma_f32_16x16x32_f16(
                false, ah, false, bl, (short)0, acc[nt], false, false);
            acc[nt] = __builtin_amdgcn_wmma_f32_16x16x32_f16(
                false, al, false, bh, (short)0, acc[nt], false, false);
        }
    }

    // ---- Segment reduction ----
    // lane (l&15) finds segment of row m0+(l&15): searchsorted(ptr, r, 'right')-1
    const long long r = (long long)(m0 + (lane & 15));
    int lo_ = 0, hi_ = nSeg + 1;
    while (lo_ < hi_) {
        const int mid = (lo_ + hi_) >> 1;
        if (ptr[mid] <= r) lo_ = mid + 1; else hi_ = mid;
    }
    const int seg = lo_ - 1;

    const int s0  = __shfl(seg, 0, 32);
    const int s15 = __shfl(seg, 15, 32);

    if (s0 == s15) {
        // Whole 16-row tile in one segment: reduce in-wave, 16 atomics per ntile.
        float* outSeg = out + (size_t)s0 * EDIM;
#pragma unroll
        for (int nt = 0; nt < 8; ++nt) {
            float p = 0.0f;
#pragma unroll
            for (int j = 0; j < 8; ++j) p += acc[nt][j];   // 8 rows in this half-wave
            p += __shfl_xor(p, 16, 32);                    // + other 8 rows
            p += 16.0f * b[nt * 16 + (lane & 15)];         // bias once per node
            if (lane < 16) atomicAdd(outSeg + nt * 16 + lane, p);
        }
    } else {
        // Tile spans a segment boundary: per-row atomics.
#pragma unroll
        for (int nt = 0; nt < 8; ++nt) {
            const int col = nt * 16 + (lane & 15);
            const float bb = b[col];
#pragma unroll
            for (int j = 0; j < 8; ++j) {
                // VGPR j holds row j (lanes 0-15) / row j+8 (lanes 16-31)
                const int rsel = j + ((lane & 16) >> 1);
                const int sj   = __shfl(seg, rsel, 32);
                atomicAdd(out + (size_t)sj * EDIM + col, acc[nt][j] + bb);
            }
        }
    }
}

extern "C" void kernel_launch(void* const* d_in, const int* in_sizes, int n_in,
                              void* d_out, int out_size, void* d_ws, size_t ws_size,
                              hipStream_t stream) {
    const float*     h_node = (const float*)d_in[0];      // [N,128] f32
    const float*     x      = (const float*)d_in[1];      // [N,128] f32
    const long long* ptr    = (const long long*)d_in[2];  // [G+1] i64
    const float*     W      = (const float*)d_in[3];      // [128,256] f32
    const float*     b      = (const float*)d_in[4];      // [128] f32
    float*           out    = (float*)d_out;              // [G,128] f32

    const int N = in_sizes[0] / EDIM;
    const int G = in_sizes[2] - 1;

    // atomics accumulate into out -> zero it every call (graph-capture safe)
    hipMemsetAsync(d_out, 0, (size_t)out_size * sizeof(float), stream);

    const int grid = (N + ROWS_PER_BLOCK - 1) / ROWS_PER_BLOCK;
    dag_encoder_wmma_kernel<<<grid, BLOCK_THREADS, 0, stream>>>(
        h_node, x, ptr, W, b, out, N, G);
}
